// CapsuleNet_30897994728314
// MI455X (gfx1250) — matrix operand order repaired and checked by reading, not compiled, not run on hardware
//
#include <hip/hip_runtime.h>
#include <hip/hip_bf16.h>

// CapsuleNet forward for MI455X (gfx1250, wave32, WMMA).
// All GEMMs use v_wmma_f32_16x16x32_f16. Fragments are built directly from
// global memory: A uses contiguous k (k=16h+s, as B does), so it reads row-major f16
// rows (two b128 loads/lane), and B fragments are per-column gathers with
// inline f32->f16 cvt. Conv operands are zero-padded once (w: K 243->256,
// x: 3->4 channels) so the implicit-GEMM inner loop is completely guard-free.
// Workspace requirement: ~139 MB (f16 conv activation dominates).

typedef __attribute__((ext_vector_type(16))) _Float16 v16h;
typedef __attribute__((ext_vector_type(8)))  _Float16 v8h;
typedef __attribute__((ext_vector_type(8)))  float    v8f;

#define DEV static __device__ __forceinline__

// ---------------- fragment helpers (wave32 WMMA f16 layouts) ----------------

// A fragment (16x32 f16) directly from row-major f16, lda in elements.
// lane<16: row=lane, K {0..15}; lane>=16: same row, K {16..31}.
DEV v16h a_frag_f16(const _Float16* __restrict__ A, int lda, int lane) {
  const int row  = lane & 15;
  const int koff = (lane >> 4) << 4;
  const _Float16* p = A + (long)row * lda + koff;
  v8h lo = *(const v8h*)(p);
  v8h hi = *(const v8h*)(p + 8);
  return __builtin_shufflevector(lo, hi, 0,1,2,3,4,5,6,7,8,9,10,11,12,13,14,15);
}

// B fragment (32x16) from row-major f32 [K][N]; B points at (k0,n0).
// lane: column n=lane&15, K range (lane>>4)*16 .. +16 (contiguous halves).
DEV v16h b_frag_f32(const float* __restrict__ B, int ldb, int lane) {
  const int n  = lane & 15;
  const int kb = (lane >> 4) << 4;
  v16h r;
#pragma unroll
  for (int j = 0; j < 16; ++j)
    r[j] = (_Float16)B[(long)(kb + j) * ldb + n];
  return r;
}

DEV v8f wmma32(v16h a, v16h b, v8f c) {
  return __builtin_amdgcn_wmma_f32_16x16x32_f16(false, a, false, b,
                                                (short)0, c, false, false);
}

// ---------------- workspace layout (bytes) ----------------
#define OFF_H       0UL                     // h f16 [256][256][1024]           = 134217728
#define OFF_UPART   134217728UL             // split-K partials [32][32768] f32 = 4194304
#define OFF_USQ     138412032UL             // squashed u f32 [256][128]        = 131072
#define OFF_PRIORS  138543104UL             // priors f32 [10][256][16][16]     = 2621440
#define OFF_V       141164544UL             // v f32 [256][10][16]              = 163840
#define OFF_D0      141328384UL             // masked caps f16 [256][160]       = 81920
#define OFF_D1      141410304UL             // d1 f16 [256][512]                = 262144
#define OFF_D2      141672448UL             // d2 f16 [256][1024]               = 524288
#define OFF_W16     142196736UL             // conv_w f16 padded [256][256]     = 131072
#define OFF_XPAD    142327808UL             // x f16 padded [256][4][40][40]    = 3276800
// total ~145604608 bytes (~139 MB)

// ---------------- prep: conv_w f32[256][243] -> f16[256][256] zero-padded ---
__global__ void prep_convw_kernel(const float* __restrict__ conv_w,
                                  _Float16* __restrict__ w16) {
  const int idx = blockIdx.x * 256 + threadIdx.x;   // < 65536
  const int oc = idx >> 8, k = idx & 255;
  w16[idx] = (k < 243) ? (_Float16)conv_w[oc * 243 + k] : (_Float16)0.f;
}

// ---------------- prep: x f32[256][3][40][40] -> f16[256][4][40][40] --------
__global__ void prep_x_kernel(const float* __restrict__ x,
                              _Float16* __restrict__ xpad) {
  const int idx = blockIdx.x * 256 + threadIdx.x;   // < 1638400
  const int b = idx / 6400;
  const int rem = idx % 6400;
  const int ci = rem / 1600, pos = rem % 1600;
  xpad[idx] = (ci < 3) ? (_Float16)x[(b * 3 + ci) * 1600 + pos] : (_Float16)0.f;
}

// ---------------- conv1 (implicit GEMM) + bias + relu -> h f16 -------------
// M=oc(256), N=spatial(1024 per b), Kpad=256. Guard-free: padded operands,
// k->(ci,r,q) address math hoisted into a per-block LDS offset table.
__global__ void __launch_bounds__(128) conv_kernel(
    const _Float16* __restrict__ xpad, const _Float16* __restrict__ w16,
    const float* __restrict__ conv_b, _Float16* __restrict__ h16) {
  __shared__ int koff[256];
  for (int t = threadIdx.x; t < 256; t += 128) {
    const int ci = t / 81, rem = t % 81;
    koff[t] = ci * 1600 + (rem / 9) * 40 + (rem % 9);
  }
  __syncthreads();
  const int wave = threadIdx.x >> 5;
  const int lane = threadIdx.x & 31;
  const int oc0  = blockIdx.y * 16;
  const int b    = blockIdx.z;
  const int s    = blockIdx.x * 64 + wave * 16 + (lane & 15);
  const int kboff = (lane >> 4) << 4;
  const _Float16* xb = xpad + (long)b * 6400 + (s >> 5) * 40 + (s & 31);
  v8f acc = {};
  for (int k0 = 0; k0 < 256; k0 += 32) {
    v16h a = a_frag_f16(w16 + oc0 * 256 + k0, 256, lane);
    int offs[16];
#pragma unroll
    for (int j = 0; j < 16; ++j) offs[j] = koff[k0 + kboff + j];
    v16h bf;
#pragma unroll
    for (int j = 0; j < 16; ++j) bf[j] = xb[offs[j]];
    acc = wmma32(a, bf, acc);
  }
  const int mrow = (lane >> 4) << 3;
#pragma unroll
  for (int r = 0; r < 8; ++r) {
    const int oc = oc0 + r + mrow;
    float val = acc[r] + conv_b[oc];
    val = fmaxf(val, 0.f);
    h16[((long)(b * 256 + oc)) * 1024 + s] = (_Float16)val;
  }
}

// ---------------- primary caps GEMM: u = h[256,262144] @ pc_w[262144,128] ---
// split-K: grid (ntile 8, mtile 16, kz 32); 8 waves split kz chunk further.
// Deterministic: block-reduced partials written to upart[kz], reduced later.
__global__ void __launch_bounds__(256) pc_gemm_kernel(
    const _Float16* __restrict__ h16, const float* __restrict__ pc_w,
    float* __restrict__ upart) {
  __shared__ float red[8 * 256];
  const int wave = threadIdx.x >> 5;
  const int lane = threadIdx.x & 31;
  const int m0 = blockIdx.y * 16, n0 = blockIdx.x * 16;
  const long kbase = (long)blockIdx.z * 8192 + (long)wave * 1024;
  v8f acc = {};
  for (int kk = 0; kk < 1024; kk += 32) {
    const long k0 = kbase + kk;
    // stream hint for the 134MB pc_w scan (global_prefetch_b8)
    if (kk + 256 < 1024)
      __builtin_prefetch(pc_w + (k0 + 256) * 128 + n0, 0, 1);
    v16h a  = a_frag_f16(h16 + (long)m0 * 262144 + k0, 262144, lane);
    v16h bf = b_frag_f32(pc_w + k0 * 128 + n0, 128, lane);
    acc = wmma32(a, bf, acc);
  }
  const int mrow = (lane >> 4) << 3;
#pragma unroll
  for (int r = 0; r < 8; ++r)
    red[wave * 256 + (r + mrow) * 16 + (lane & 15)] = acc[r];
  __syncthreads();
  float s = 0.f;
#pragma unroll
  for (int w = 0; w < 8; ++w) s += red[w * 256 + threadIdx.x];
  const int m  = m0 + (int)(threadIdx.x >> 4);
  const int nn = n0 + (int)(threadIdx.x & 15);
  upart[(long)blockIdx.z * 32768 + m * 128 + nn] = s;
}

// ---------------- reduce split-K, add bias, squash -> usq -------------------
__global__ void squash_u_kernel(const float* __restrict__ upart,
                                const float* __restrict__ pc_b,
                                float* __restrict__ usq) {
  const int t = blockIdx.x * blockDim.x + threadIdx.x;   // 0..4095
  if (t >= 4096) return;
  const int b = t >> 4, p = t & 15;
  float vv[8];
  float sn = 0.f;
#pragma unroll
  for (int i = 0; i < 8; ++i) {
    const int idx = b * 128 + p * 8 + i;
    float val = pc_b[p * 8 + i];
    for (int z = 0; z < 32; ++z) val += upart[(long)z * 32768 + idx];
    vv[i] = val;
    sn += val * val;
  }
  const float scale = (sn / (1.f + sn)) * rsqrtf(sn);
#pragma unroll
  for (int i = 0; i < 8; ++i) usq[b * 128 + p * 8 + i] = vv[i] * scale;
}

// ---------------- priors: [c][b][p][o] = sum_i usq[b,p,i]*route_w[c,p,i,o] --
__global__ void priors_kernel(const float* __restrict__ usq,
                              const float* __restrict__ route_w,
                              float* __restrict__ priors) {
  const int idx = blockIdx.x * 256 + threadIdx.x;   // < 655360
  const int o = idx & 15;
  const int p = (idx >> 4) & 15;
  const int b = (idx >> 8) & 255;
  const int c = idx >> 16;
  float s = 0.f;
#pragma unroll
  for (int i = 0; i < 8; ++i)
    s += usq[b * 128 + p * 8 + i] * route_w[((c * 16 + p) * 8 + i) * 16 + o];
  priors[idx] = s;
}

// ---------------- dynamic routing, one wave per (c,b) pair ------------------
__global__ void __launch_bounds__(256) routing_kernel(
    const float* __restrict__ priors, float* __restrict__ v) {
  __shared__ float lds[8 * 320];
  const int wave = threadIdx.x >> 5, lane = threadIdx.x & 31;
  float* pr   = &lds[wave * 320];   // [16 p][16 o]
  float* lg   = pr + 256;           // logits[16]
  float* prob = lg + 16;            // probs[16]
  float* sv   = prob + 16;          // s/out[16]
  float* red  = sv + 16;            // scalars
  const int pair = blockIdx.x * 8 + wave;   // c*256 + b
  const int c = pair >> 8, b = pair & 255;
  const float* P = priors + (long)pair * 256;
#pragma unroll
  for (int j = 0; j < 8; ++j) pr[lane * 8 + j] = P[lane * 8 + j];
  if (lane < 16) lg[lane] = 0.f;
  __syncthreads();
  for (int it = 0; it < 3; ++it) {
    if (lane == 0) {
      float m = lg[0];
      for (int p = 1; p < 16; ++p) m = fmaxf(m, lg[p]);
      red[0] = m;
    }
    __syncthreads();
    if (lane < 16) prob[lane] = __expf(lg[lane] - red[0]);
    __syncthreads();
    if (lane == 0) {
      float ssum = 0.f;
      for (int p = 0; p < 16; ++p) ssum += prob[p];
      red[1] = 1.f / ssum;
    }
    __syncthreads();
    if (lane < 16) {
      float a = 0.f;
      for (int p = 0; p < 16; ++p) a += prob[p] * red[1] * pr[p * 16 + lane];
      sv[lane] = a;
    }
    __syncthreads();
    if (lane == 0) {
      float sn = 0.f;
      for (int o = 0; o < 16; ++o) sn += sv[o] * sv[o];
      red[2] = (sn / (1.f + sn)) * rsqrtf(sn);
    }
    __syncthreads();
    if (lane < 16) sv[lane] *= red[2];          // out = squash(s)
    __syncthreads();
    if (it < 2 && lane < 16) {
      float d = 0.f;
      for (int o = 0; o < 16; ++o) d += pr[lane * 16 + o] * sv[o];
      lg[lane] += d;
    }
    __syncthreads();
  }
  if (lane < 16) v[(b * 10 + c) * 16 + lane] = sv[lane];
}

// ---------------- class softmax + label masking -> classes, d0(f16) --------
__global__ void head_kernel(const float* __restrict__ v, const float* __restrict__ y,
                            float* __restrict__ classes, _Float16* __restrict__ d0) {
  const int b = threadIdx.x;   // 256 threads, 1 block
  float norms[10];
  float m = -1e30f;
#pragma unroll
  for (int c = 0; c < 10; ++c) {
    float sn = 0.f;
    for (int o = 0; o < 16; ++o) { float t = v[(b * 10 + c) * 16 + o]; sn += t * t; }
    norms[c] = sqrtf(sn);
    m = fmaxf(m, norms[c]);
  }
  float ssum = 0.f;
#pragma unroll
  for (int c = 0; c < 10; ++c) { norms[c] = __expf(norms[c] - m); ssum += norms[c]; }
#pragma unroll
  for (int c = 0; c < 10; ++c) classes[b * 10 + c] = norms[c] / ssum;
#pragma unroll
  for (int c = 0; c < 10; ++c) {
    const float yc = y[b * 10 + c];
    for (int o = 0; o < 16; ++o)
      d0[b * 160 + c * 16 + o] = (_Float16)(v[(b * 10 + c) * 16 + o] * yc);
  }
}

// ---------------- decoder GEMMs: out = act(A@B + bias) ----------------------
// ACT==0: relu -> f16 out ; ACT==1: sigmoid -> f32 out. One wave per 16x16 tile.
template<int ACT>
__global__ void __launch_bounds__(256) dec_gemm_kernel(
    const _Float16* __restrict__ A, int lda,
    const float* __restrict__ B, int ldb,
    const float* __restrict__ bias, int K, int ntiles,
    _Float16* __restrict__ out16, float* __restrict__ out32, int ldo) {
  const int wave = threadIdx.x >> 5, lane = threadIdx.x & 31;
  const int nt = blockIdx.x * 8 + wave;
  if (nt >= ntiles) return;
  const int n0 = nt * 16, m0 = blockIdx.y * 16;
  v8f acc = {};
  for (int k0 = 0; k0 < K; k0 += 32) {
    v16h a  = a_frag_f16(A + (long)m0 * lda + k0, lda, lane);
    v16h bf = b_frag_f32(B + (long)k0 * ldb + n0, ldb, lane);
    acc = wmma32(a, bf, acc);
  }
  const int n = n0 + (lane & 15);
  const int mrow = (lane >> 4) << 3;
  const float bn = bias[n];
#pragma unroll
  for (int r = 0; r < 8; ++r) {
    const int m = m0 + r + mrow;
    float val = acc[r] + bn;
    if (ACT == 0) {
      val = fmaxf(val, 0.f);
      out16[(long)m * ldo + n] = (_Float16)val;
    } else {
      val = 1.f / (1.f + __expf(-val));
      out32[(long)m * ldo + n] = val;
    }
  }
}

// ---------------- launch ----------------------------------------------------
extern "C" void kernel_launch(void* const* d_in, const int* in_sizes, int n_in,
                              void* d_out, int out_size, void* d_ws, size_t ws_size,
                              hipStream_t stream) {
  (void)in_sizes; (void)n_in; (void)out_size; (void)ws_size;
  const float* x       = (const float*)d_in[0];
  const float* y       = (const float*)d_in[1];
  const float* conv_w  = (const float*)d_in[2];
  const float* conv_b  = (const float*)d_in[3];
  const float* pc_w    = (const float*)d_in[4];
  const float* pc_b    = (const float*)d_in[5];
  const float* route_w = (const float*)d_in[6];
  const float* w1 = (const float*)d_in[7];
  const float* b1 = (const float*)d_in[8];
  const float* w2 = (const float*)d_in[9];
  const float* b2 = (const float*)d_in[10];
  const float* w3 = (const float*)d_in[11];
  const float* b3 = (const float*)d_in[12];

  char* ws = (char*)d_ws;
  _Float16* h16   = (_Float16*)(ws + OFF_H);
  float*    upart = (float*)(ws + OFF_UPART);
  float*    usq   = (float*)(ws + OFF_USQ);
  float*    priors= (float*)(ws + OFF_PRIORS);
  float*    v     = (float*)(ws + OFF_V);
  _Float16* d0    = (_Float16*)(ws + OFF_D0);
  _Float16* d1    = (_Float16*)(ws + OFF_D1);
  _Float16* d2    = (_Float16*)(ws + OFF_D2);
  _Float16* w16   = (_Float16*)(ws + OFF_W16);
  _Float16* xpad  = (_Float16*)(ws + OFF_XPAD);
  float* classes = (float*)d_out;
  float* recon   = (float*)d_out + 2560;

  prep_convw_kernel<<<256, 256, 0, stream>>>(conv_w, w16);
  prep_x_kernel<<<6400, 256, 0, stream>>>(x, xpad);
  conv_kernel<<<dim3(16, 16, 256), 128, 0, stream>>>(xpad, w16, conv_b, h16);
  pc_gemm_kernel<<<dim3(8, 16, 32), 256, 0, stream>>>(h16, pc_w, upart);
  squash_u_kernel<<<16, 256, 0, stream>>>(upart, pc_b, usq);
  priors_kernel<<<2560, 256, 0, stream>>>(usq, route_w, priors);
  routing_kernel<<<320, 256, 0, stream>>>(priors, v);
  head_kernel<<<1, 256, 0, stream>>>(v, y, classes, d0);
  dec_gemm_kernel<0><<<dim3(4, 16), 256, 0, stream>>>(d0, 160, w1, 512, b1, 160, 32, d1, nullptr, 512);
  dec_gemm_kernel<0><<<dim3(8, 16), 256, 0, stream>>>(d1, 512, w2, 1024, b2, 512, 64, d2, nullptr, 1024);
  dec_gemm_kernel<1><<<dim3(13, 16), 256, 0, stream>>>(d2, 1024, w3, 1600, b3, 1024, 100, nullptr, recon, 1600);
}
